// LayerNormLSTMCell_10608569221488
// MI455X (gfx1250) — matrix operand order repaired and checked by reading, not compiled, not run on hardware
//
#include <hip/hip_runtime.h>
#include <hip/hip_bf16.h>
#include <stdint.h>

// ---------------------------------------------------------------------------
// LayerNorm LSTM cell for MI455X (gfx1250, wave32, WMMA, async LDS DMA).
//
// Phase 0: one-time fp32 -> bf16 hi/lo split of x, h_prev (row-major) and
//          W_ih, W_hh (transposed to [N][K]) so the GEMM hot loop has zero
//          conversion VALU and both tiles are row-contiguous for async copy.
// Phase 1: bf16x3-split GEMM (fp32 emulation) with double-buffered
//          GLOBAL_LOAD_ASYNC_TO_LDS_B128 staging + v_wmma_f32_16x16x32_bf16.
// Phase 2: fused row-wise LayerNorm(ih)+LayerNorm(hh)+LSTM+LayerNorm(c_t).
// ---------------------------------------------------------------------------

typedef __bf16  v8bf  __attribute__((ext_vector_type(8)));
typedef __bf16  v16bf __attribute__((ext_vector_type(16)));
typedef float   v8f   __attribute__((ext_vector_type(8)));

#define B_DIM 16384
#define K_DIM 512
#define N_DIM 2048   // 4*H
#define H_DIM 512

#define MT 128       // workgroup M tile
#define NT 128       // workgroup N tile
#define KT 32        // K step (one WMMA depth)
#define LDSP 40      // padded LDS row stride in bf16 elements (80B, 16B aligned)
#define KSTEPS (K_DIM / KT)

// ---- async global->LDS copy (CDNA5) with synchronous fallback --------------
#if defined(__HIP_DEVICE_COMPILE__) && __has_builtin(__builtin_amdgcn_global_load_async_to_lds_b128)
#define USE_ASYNC_LDS 1
#else
#define USE_ASYNC_LDS 0
#endif

#if USE_ASYNC_LDS
// Builtin parameter type (from hipcc diagnostic): pointer to
// __attribute__((vector_size(16))) int, in global / LDS address spaces.
typedef int v4i_vs __attribute__((vector_size(16)));
typedef __attribute__((address_space(1))) v4i_vs as1_v4i;
typedef __attribute__((address_space(3))) v4i_vs as3_v4i;
#if __has_builtin(__builtin_amdgcn_s_wait_asynccnt)
#define WAIT_ASYNC(n) __builtin_amdgcn_s_wait_asynccnt(n)
#else
#define WAIT_ASYNC(n) asm volatile("s_wait_asynccnt %0" ::"i"(n))
#endif
#else
#define WAIT_ASYNC(n)
#endif

__device__ __forceinline__ void async_cp16(const __bf16* g, __bf16* l) {
#if USE_ASYNC_LDS
    // Generic LDS pointers carry the LDS byte offset in their low 32 bits
    // (see ISA section 10.2); as1 pointer is the full VA.
    __builtin_amdgcn_global_load_async_to_lds_b128(
        (as1_v4i*)(uintptr_t)g,
        (as3_v4i*)(uint32_t)(uintptr_t)l,
        0, 0);
#else
    *(v8bf*)l = *(const v8bf*)g;
#endif
}

// Build a v16bf fragment from two 8-element LDS chunks.
__device__ __forceinline__ v16bf frag_from2(const __bf16* p0, const __bf16* p1) {
    v8bf a = *(const v8bf*)p0;
    v8bf b = *(const v8bf*)p1;
    v16bf r;
#pragma unroll
    for (int i = 0; i < 8; ++i) { r[i] = a[i]; r[i + 8] = b[i]; }
    return r;
}

// ---------------------------------------------------------------------------
// Phase 0a: split activations to bf16 hi/lo (same layout, [B][K] row-major).
// ---------------------------------------------------------------------------
__global__ __launch_bounds__(256)
void split_a(const float* __restrict__ x, const float* __restrict__ h_prev,
             __bf16* __restrict__ xhi, __bf16* __restrict__ xlo,
             __bf16* __restrict__ hhi, __bf16* __restrict__ hlo) {
    const float* src = blockIdx.y ? h_prev : x;
    __bf16* hi = blockIdx.y ? hhi : xhi;
    __bf16* lo = blockIdx.y ? hlo : xlo;

    const size_t base = ((size_t)blockIdx.x * 256 + threadIdx.x) * 8;
    const float4* p = (const float4*)(src + base);
    float4 f0 = p[0], f1 = p[1];
    float vals[8] = {f0.x, f0.y, f0.z, f0.w, f1.x, f1.y, f1.z, f1.w};
    v8bf h8, l8;
#pragma unroll
    for (int i = 0; i < 8; ++i) {
        __bf16 h = (__bf16)vals[i];
        h8[i] = h;
        l8[i] = (__bf16)(vals[i] - (float)h);
    }
    *(v8bf*)(hi + base) = h8;
    *(v8bf*)(lo + base) = l8;
}

// ---------------------------------------------------------------------------
// Phase 0b: transpose + split weights: [K][N] fp32 -> [N][K] bf16 hi/lo.
// ---------------------------------------------------------------------------
__global__ __launch_bounds__(256)
void split_transpose_w(const float* __restrict__ W_ih, const float* __restrict__ W_hh,
                       __bf16* __restrict__ tih_hi, __bf16* __restrict__ tih_lo,
                       __bf16* __restrict__ thh_hi, __bf16* __restrict__ thh_lo) {
    const float* W = blockIdx.z ? W_hh : W_ih;
    __bf16* thi = blockIdx.z ? thh_hi : tih_hi;
    __bf16* tlo = blockIdx.z ? thh_lo : tih_lo;

    __shared__ float tile[32][33];
    const int k0 = blockIdx.x * 32;     // K block
    const int n0 = blockIdx.y * 32;     // N block
    const int t  = threadIdx.x;
    const int tn = t & 31;
    const int tg = t >> 5;              // 0..7

#pragma unroll
    for (int r = 0; r < 4; ++r) {
        const int k = tg * 4 + r;
        tile[k][tn] = W[(size_t)(k0 + k) * N_DIM + n0 + tn];
    }
    __syncthreads();
#pragma unroll
    for (int r = 0; r < 4; ++r) {
        const int n = tg * 4 + r;
        const float v = tile[tn][n];
        const __bf16 h = (__bf16)v;
        thi[(size_t)(n0 + n) * K_DIM + k0 + tn] = h;
        tlo[(size_t)(n0 + n) * K_DIM + k0 + tn] = (__bf16)(v - (float)h);
    }
}

// ---------------------------------------------------------------------------
// Phase 1: bf16x3 GEMM with double-buffered async LDS staging.
// A tiles (bf16 [B][K]) and Bt tiles (bf16 [N][K]) are both 128 rows x 64B.
// ---------------------------------------------------------------------------
__device__ __forceinline__ void stage_tile(const __bf16* __restrict__ Ahg,
                                           const __bf16* __restrict__ Alg,
                                           const __bf16* __restrict__ Bhg,
                                           const __bf16* __restrict__ Blg,
                                           int m0, int n0, int k0, int tid,
                                           __bf16* Ah, __bf16* Al,
                                           __bf16* Bh, __bf16* Bl) {
#pragma unroll
    for (int c = 0; c < 2; ++c) {
        const int chunk = tid + c * 256;     // 0..511 : row*4 + 16B-quarter
        const int row = chunk >> 2;
        const int q   = chunk & 3;
        const size_t goffA = (size_t)(m0 + row) * K_DIM + k0 + q * 8;
        const size_t goffB = (size_t)(n0 + row) * K_DIM + k0 + q * 8;
        const int    loff  = row * LDSP + q * 8;
        async_cp16(Ahg + goffA, Ah + loff);
        async_cp16(Alg + goffA, Al + loff);
        async_cp16(Bhg + goffB, Bh + loff);
        async_cp16(Blg + goffB, Bl + loff);
    }
}

__global__ __launch_bounds__(256)
void lnlstm_gemm_bf16x3(const __bf16* __restrict__ xhi, const __bf16* __restrict__ xlo,
                        const __bf16* __restrict__ hhi, const __bf16* __restrict__ hlo,
                        const __bf16* __restrict__ tih_hi, const __bf16* __restrict__ tih_lo,
                        const __bf16* __restrict__ thh_hi, const __bf16* __restrict__ thh_lo,
                        float* __restrict__ gates_ih, float* __restrict__ gates_hh) {
    const __bf16* Ahg = (blockIdx.z == 0) ? xhi : hhi;
    const __bf16* Alg = (blockIdx.z == 0) ? xlo : hlo;
    const __bf16* Bhg = (blockIdx.z == 0) ? tih_hi : thh_hi;
    const __bf16* Blg = (blockIdx.z == 0) ? tih_lo : thh_lo;
    float*        out = (blockIdx.z == 0) ? gates_ih : gates_hh;

    const int m0 = blockIdx.x * MT;
    const int n0 = blockIdx.y * NT;

    const int tid  = threadIdx.x;
    const int lane = tid & 31;
    const int wave = tid >> 5;
    const int wm   = wave >> 1;          // 0..3 : 32-row group
    const int wn   = wave & 1;           // 0..1 : 64-col group
    const int lr   = lane & 15;
    const int lh   = lane >> 4;

    __shared__ __attribute__((aligned(16))) __bf16 Ah[2][MT * LDSP];
    __shared__ __attribute__((aligned(16))) __bf16 Al[2][MT * LDSP];
    __shared__ __attribute__((aligned(16))) __bf16 Bh[2][NT * LDSP];
    __shared__ __attribute__((aligned(16))) __bf16 Bl[2][NT * LDSP];

    const v8f vzero = {0.f, 0.f, 0.f, 0.f, 0.f, 0.f, 0.f, 0.f};
    v8f acc[2][4];
#pragma unroll
    for (int mi = 0; mi < 2; ++mi)
#pragma unroll
        for (int ni = 0; ni < 4; ++ni) acc[mi][ni] = vzero;

    // Prologue: issue async loads for buffer 0 (8 b128 per wave).
    stage_tile(Ahg, Alg, Bhg, Blg, m0, n0, 0, tid, Ah[0], Al[0], Bh[0], Bl[0]);

    int cur = 0;
    for (int ks = 0; ks < KSTEPS; ++ks) {
        if (ks + 1 < KSTEPS) {
            // Issue next buffer's 8 async copies, then wait until only those
            // 8 are outstanding -> current buffer is complete for this wave.
            stage_tile(Ahg, Alg, Bhg, Blg, m0, n0, (ks + 1) * KT, tid,
                       Ah[cur ^ 1], Al[cur ^ 1], Bh[cur ^ 1], Bl[cur ^ 1]);
            WAIT_ASYNC(8);
        } else {
            WAIT_ASYNC(0);
        }
        __syncthreads();   // all waves' async writes to buf[cur] visible

        const __bf16* AhC = Ah[cur];
        const __bf16* AlC = Al[cur];
        const __bf16* BhC = Bh[cur];
        const __bf16* BlC = Bl[cur];

        // A fragment (16x32 16-bit): lane<16 -> row M=lane, K {kh8..+7, kh8+16..+23}
        v16bf aHi[2], aLo[2];
        const int kh8 = lh * 8;
#pragma unroll
        for (int mi = 0; mi < 2; ++mi) {
            const int row = wm * 32 + mi * 16 + lr;
            aHi[mi] = frag_from2(&AhC[row * LDSP + kh8], &AhC[row * LDSP + kh8 + 16]);
            aLo[mi] = frag_from2(&AlC[row * LDSP + kh8], &AlC[row * LDSP + kh8 + 16]);
        }
        // B fragment (32x16): col N = lane&15, K {kh16..+7, kh16+16..+23} (same as A)
        v16bf bHi[4], bLo[4];
        const int kh16 = lh * 8;
#pragma unroll
        for (int ni = 0; ni < 4; ++ni) {
            const int coln = wn * 64 + ni * 16 + lr;
            bHi[ni] = frag_from2(&BhC[coln * LDSP + kh16], &BhC[coln * LDSP + kh16 + 16]);
            bLo[ni] = frag_from2(&BlC[coln * LDSP + kh16], &BlC[coln * LDSP + kh16 + 16]);
        }

        // bf16x3: hi*hi + hi*lo + lo*hi (fp32 emulation)
#pragma unroll
        for (int mi = 0; mi < 2; ++mi) {
#pragma unroll
            for (int ni = 0; ni < 4; ++ni) {
                acc[mi][ni] = __builtin_amdgcn_wmma_f32_16x16x32_bf16(
                    false, aHi[mi], false, bHi[ni], (short)0, acc[mi][ni], false, false);
                acc[mi][ni] = __builtin_amdgcn_wmma_f32_16x16x32_bf16(
                    false, aHi[mi], false, bLo[ni], (short)0, acc[mi][ni], false, false);
                acc[mi][ni] = __builtin_amdgcn_wmma_f32_16x16x32_bf16(
                    false, aLo[mi], false, bHi[ni], (short)0, acc[mi][ni], false, false);
            }
        }

        __syncthreads();   // everyone done reading buf[cur] before its reuse
        cur ^= 1;
    }

    // Epilogue: C/D layout: VGPR r -> (M = r + 8*lane_half, N = lane&15)
#pragma unroll
    for (int mi = 0; mi < 2; ++mi) {
#pragma unroll
        for (int ni = 0; ni < 4; ++ni) {
#pragma unroll
            for (int r = 0; r < 8; ++r) {
                const int row = m0 + wm * 32 + mi * 16 + lh * 8 + r;
                const int col = n0 + wn * 64 + ni * 16 + lr;
                out[(size_t)row * N_DIM + col] = acc[mi][ni][r];
            }
        }
    }
}

// ---------------------------------------------------------------------------
// Phase 2: fused LayerNorm + LSTM elementwise, one 256-thread block per row.
// ---------------------------------------------------------------------------
__device__ __forceinline__ float sigmoidf_(float v) {
    return 1.0f / (1.0f + __expf(-v));
}

__global__ __launch_bounds__(256)
void lnlstm_pointwise(const float* __restrict__ gih,
                      const float* __restrict__ ghh,
                      const float* __restrict__ c_prev,
                      const float* __restrict__ gamma_ih, const float* __restrict__ beta_ih,
                      const float* __restrict__ gamma_hh, const float* __restrict__ beta_hh,
                      const float* __restrict__ gamma_c,  const float* __restrict__ beta_c,
                      float* __restrict__ out) {
    const int b = blockIdx.x;
    const int t = threadIdx.x;
    const int lane = t & 31;
    const int wv = t >> 5;

    __shared__ float gbuf[N_DIM];
    __shared__ float red[32];

    const int c0 = t * 8;
    float vih[8], vhh[8];
    {
        const float4* pih = (const float4*)(gih + (size_t)b * N_DIM + c0);
        const float4* phh = (const float4*)(ghh + (size_t)b * N_DIM + c0);
        float4 a0 = pih[0], a1 = pih[1];
        float4 b0 = phh[0], b1 = phh[1];
        vih[0]=a0.x; vih[1]=a0.y; vih[2]=a0.z; vih[3]=a0.w;
        vih[4]=a1.x; vih[5]=a1.y; vih[6]=a1.z; vih[7]=a1.w;
        vhh[0]=b0.x; vhh[1]=b0.y; vhh[2]=b0.z; vhh[3]=b0.w;
        vhh[4]=b1.x; vhh[5]=b1.y; vhh[6]=b1.z; vhh[7]=b1.w;
    }

    float s_ih = 0.f, q_ih = 0.f, s_hh = 0.f, q_hh = 0.f;
#pragma unroll
    for (int i = 0; i < 8; ++i) {
        s_ih += vih[i]; q_ih += vih[i] * vih[i];
        s_hh += vhh[i]; q_hh += vhh[i] * vhh[i];
    }
#pragma unroll
    for (int off = 16; off > 0; off >>= 1) {
        s_ih += __shfl_xor(s_ih, off, 32);
        q_ih += __shfl_xor(q_ih, off, 32);
        s_hh += __shfl_xor(s_hh, off, 32);
        q_hh += __shfl_xor(q_hh, off, 32);
    }
    if (lane == 0) {
        red[wv] = s_ih; red[8 + wv] = q_ih; red[16 + wv] = s_hh; red[24 + wv] = q_hh;
    }
    __syncthreads();
    float Sih = 0.f, Qih = 0.f, Shh = 0.f, Qhh = 0.f;
#pragma unroll
    for (int i = 0; i < 8; ++i) {
        Sih += red[i]; Qih += red[8 + i]; Shh += red[16 + i]; Qhh += red[24 + i];
    }
    const float invG = 1.0f / (float)N_DIM;
    const float m_ih = Sih * invG;
    const float m_hh = Shh * invG;
    const float r_ih = rsqrtf(Qih * invG - m_ih * m_ih + 1e-5f);
    const float r_hh = rsqrtf(Qhh * invG - m_hh * m_hh + 1e-5f);

#pragma unroll
    for (int i = 0; i < 8; ++i) {
        const int j = c0 + i;
        gbuf[j] = (vih[i] - m_ih) * r_ih * gamma_ih[j] + beta_ih[j]
                + (vhh[i] - m_hh) * r_hh * gamma_hh[j] + beta_hh[j];
    }
    __syncthreads();

    float cvals[2], ovals[2];
    float cs = 0.f, cq = 0.f;
#pragma unroll
    for (int e = 0; e < 2; ++e) {
        const int j = t + e * 256;
        const float ig = sigmoidf_(gbuf[j]);
        const float fg = sigmoidf_(gbuf[H_DIM + j]);
        const float gg = tanhf(gbuf[2 * H_DIM + j]);
        const float og = sigmoidf_(gbuf[3 * H_DIM + j]);
        const float c  = fg * c_prev[(size_t)b * H_DIM + j] + ig * gg;
        cvals[e] = c; ovals[e] = og;
        cs += c; cq += c * c;
    }
#pragma unroll
    for (int off = 16; off > 0; off >>= 1) {
        cs += __shfl_xor(cs, off, 32);
        cq += __shfl_xor(cq, off, 32);
    }
    __syncthreads();
    if (lane == 0) { red[wv] = cs; red[8 + wv] = cq; }
    __syncthreads();
    float Sc = 0.f, Qc = 0.f;
#pragma unroll
    for (int i = 0; i < 8; ++i) { Sc += red[i]; Qc += red[8 + i]; }
    const float invH = 1.0f / (float)H_DIM;
    const float m_c = Sc * invH;
    const float r_c = rsqrtf(Qc * invH - m_c * m_c + 1e-5f);

#pragma unroll
    for (int e = 0; e < 2; ++e) {
        const int j = t + e * 256;
        const float cn = (cvals[e] - m_c) * r_c * gamma_c[j] + beta_c[j];
        out[(size_t)b * H_DIM + j] = ovals[e] * tanhf(cn);                 // h_t
        out[(size_t)B_DIM * H_DIM + (size_t)b * H_DIM + j] = cvals[e];     // c_t
    }
}

// ---------------------------------------------------------------------------

extern "C" void kernel_launch(void* const* d_in, const int* in_sizes, int n_in,
                              void* d_out, int out_size, void* d_ws, size_t ws_size,
                              hipStream_t stream) {
    const float* x        = (const float*)d_in[0];
    const float* h_prev   = (const float*)d_in[1];
    const float* c_prev   = (const float*)d_in[2];
    const float* W_ih     = (const float*)d_in[3];
    const float* W_hh     = (const float*)d_in[4];
    const float* gamma_ih = (const float*)d_in[5];
    const float* beta_ih  = (const float*)d_in[6];
    const float* gamma_hh = (const float*)d_in[7];
    const float* beta_hh  = (const float*)d_in[8];
    const float* gamma_c  = (const float*)d_in[9];
    const float* beta_c   = (const float*)d_in[10];
    float* out = (float*)d_out;

    // Workspace layout (bump allocation).
    const size_t BG = (size_t)B_DIM * N_DIM;   // 33.55 M
    const size_t BK = (size_t)B_DIM * K_DIM;   //  8.39 M
    const size_t WK = (size_t)N_DIM * K_DIM;   //  1.05 M
    char* ws = (char*)d_ws;
    float* gih = (float*)ws;  ws += BG * sizeof(float);
    float* ghh = (float*)ws;  ws += BG * sizeof(float);
    __bf16* xhi = (__bf16*)ws; ws += BK * sizeof(__bf16);
    __bf16* xlo = (__bf16*)ws; ws += BK * sizeof(__bf16);
    __bf16* hhi = (__bf16*)ws; ws += BK * sizeof(__bf16);
    __bf16* hlo = (__bf16*)ws; ws += BK * sizeof(__bf16);
    __bf16* tih_hi = (__bf16*)ws; ws += WK * sizeof(__bf16);
    __bf16* tih_lo = (__bf16*)ws; ws += WK * sizeof(__bf16);
    __bf16* thh_hi = (__bf16*)ws; ws += WK * sizeof(__bf16);
    __bf16* thh_lo = (__bf16*)ws; ws += WK * sizeof(__bf16);

    // Phase 0: precision split (+ weight transpose).
    dim3 aGrid((unsigned)(BK / (256 * 8)), 2, 1);
    split_a<<<aGrid, 256, 0, stream>>>(x, h_prev, xhi, xlo, hhi, hlo);
    dim3 wGrid(K_DIM / 32, N_DIM / 32, 2);
    split_transpose_w<<<wGrid, 256, 0, stream>>>(W_ih, W_hh,
                                                 tih_hi, tih_lo, thh_hi, thh_lo);

    // Phase 1: both GEMMs (z selects ih / hh).
    dim3 gGrid(B_DIM / MT, N_DIM / NT, 2);
    lnlstm_gemm_bf16x3<<<gGrid, 256, 0, stream>>>(xhi, xlo, hhi, hlo,
                                                  tih_hi, tih_lo, thh_hi, thh_lo,
                                                  gih, ghh);

    // Phase 2: fused LN + LSTM.
    lnlstm_pointwise<<<B_DIM, 256, 0, stream>>>(gih, ghh, c_prev,
                                                gamma_ih, beta_ih,
                                                gamma_hh, beta_hh,
                                                gamma_c, beta_c, out);
}
